// AttentionLayer_15144054686159
// MI455X (gfx1250) — compile-verified, matrix operand order checked by reading, not run on hardware
//
#include <hip/hip_runtime.h>
#include <hip/hip_bf16.h>

typedef __attribute__((ext_vector_type(16))) _Float16 v16h;
typedef __attribute__((ext_vector_type(8)))  _Float16 v8h;
typedef __attribute__((ext_vector_type(4)))  _Float16 v4h;
typedef __attribute__((ext_vector_type(8)))  float    v8f;

#define BB 4
#define SS 1024
#define DD 1024
#define NHH 16
#define DHH 64
#define DII 4096

// ISA 7.12.2 16-bit A/B lane layout: lane half hl holds two contiguous
// 8-element K-runs: [hl*8, hl*8+8) and [16+hl*8, 16+hl*8+8).
// With the K dimension contiguous in LDS, a fragment is two b128 loads.
__device__ __forceinline__ v16h frag_ld_lds(const _Float16* rowp, int hl) {
    v8h lo = *(const v8h*)(rowp + hl * 8);
    v8h hi = *(const v8h*)(rowp + 16 + hl * 8);
    return __builtin_shufflevector(lo, hi, 0, 1, 2, 3, 4, 5, 6, 7,
                                   8, 9, 10, 11, 12, 13, 14, 15);
}

// Fragment from a row-contiguous fp32 global row: 4x global_load_b128 + cvt.
__device__ __forceinline__ v16h frag_ld_g(const float* rowp, int c, int hl) {
    const float* p = rowp + c * 32 + hl * 8;
    float4 x0 = *(const float4*)(p + 0);
    float4 x1 = *(const float4*)(p + 4);
    float4 y0 = *(const float4*)(p + 16);
    float4 y1 = *(const float4*)(p + 20);
    v16h r;
    r[0]  = (_Float16)x0.x; r[1]  = (_Float16)x0.y;
    r[2]  = (_Float16)x0.z; r[3]  = (_Float16)x0.w;
    r[4]  = (_Float16)x1.x; r[5]  = (_Float16)x1.y;
    r[6]  = (_Float16)x1.z; r[7]  = (_Float16)x1.w;
    r[8]  = (_Float16)y0.x; r[9]  = (_Float16)y0.y;
    r[10] = (_Float16)y0.z; r[11] = (_Float16)y0.w;
    r[12] = (_Float16)y1.x; r[13] = (_Float16)y1.y;
    r[14] = (_Float16)y1.z; r[15] = (_Float16)y1.w;
    return r;
}

// ---------------------------------------------------------------------------
// Generic GEMM: C[M,N] = A[M,K] * B[K,N] (+bias, +relu). fp32 in/out,
// f16 WMMA. Block = 128 threads (4 waves), tile 64x64, K-step 32.
// LDS: A row-major [m][k], B transposed [n][k] so fragments are b128 loads.
// Software-pipelined: next K-tile prefetched into VGPRs during WMMA compute.
// ---------------------------------------------------------------------------
__global__ __launch_bounds__(128) void gemm_wmma_kernel(
    const float* __restrict__ A, const float* __restrict__ Bm,
    float* __restrict__ C, int M, int N, int K,
    const float* __restrict__ bias, int relu)
{
    __shared__ _Float16 sA[64 * 32] __attribute__((aligned(16)));   // [m][k]
    __shared__ _Float16 sBT[64 * 32] __attribute__((aligned(16)));  // [n][k]

    const int tid  = threadIdx.x;
    const int wave = tid >> 5;
    const int lane = tid & 31;
    const int hl   = lane >> 4;
    const int nn   = lane & 15;
    const int row0 = blockIdx.y * 64;
    const int col0 = blockIdx.x * 64;

    v8f acc[4] = {};
    float4 ra[4], rb[4];

    // prefetch first K-tile into registers
    #pragma unroll
    for (int it = 0; it < 4; it++) {
        const int e  = tid + it * 128;
        const int rm = e >> 3, rc = (e & 7) * 4;
        const int rk = e >> 4, rn = (e & 15) * 4;
        ra[it] = *(const float4*)&A[(size_t)(row0 + rm) * K + rc];
        rb[it] = *(const float4*)&Bm[(size_t)rk * N + col0 + rn];
    }

    for (int k0 = 0; k0 < K; k0 += 32) {
        // convert + store the prefetched tile to LDS
        #pragma unroll
        for (int it = 0; it < 4; it++) {
            const int e  = tid + it * 128;
            const int rm = e >> 3, rc = (e & 7) * 4;
            v4h pk;
            pk[0] = (_Float16)ra[it].x; pk[1] = (_Float16)ra[it].y;
            pk[2] = (_Float16)ra[it].z; pk[3] = (_Float16)ra[it].w;
            *(v4h*)&sA[rm * 32 + rc] = pk;
            const int rk = e >> 4, rn = (e & 15) * 4;
            sBT[(rn + 0) * 32 + rk] = (_Float16)rb[it].x;
            sBT[(rn + 1) * 32 + rk] = (_Float16)rb[it].y;
            sBT[(rn + 2) * 32 + rk] = (_Float16)rb[it].z;
            sBT[(rn + 3) * 32 + rk] = (_Float16)rb[it].w;
        }
        __syncthreads();

        // prefetch next K-tile while WMMAs run
        const int kn = k0 + 32;
        if (kn < K) {
            #pragma unroll
            for (int it = 0; it < 4; it++) {
                const int e  = tid + it * 128;
                const int rm = e >> 3, rc = (e & 7) * 4;
                const int rk = e >> 4, rn = (e & 15) * 4;
                ra[it] = *(const float4*)&A[(size_t)(row0 + rm) * K + kn + rc];
                rb[it] = *(const float4*)&Bm[(size_t)(kn + rk) * N + col0 + rn];
            }
        }

        v16h a = frag_ld_lds(&sA[(wave * 16 + nn) * 32], hl);
        #pragma unroll
        for (int t = 0; t < 4; t++) {
            v16h bf = frag_ld_lds(&sBT[(t * 16 + nn) * 32], hl);
            acc[t] = __builtin_amdgcn_wmma_f32_16x16x32_f16(
                false, a, false, bf, (short)0, acc[t], false, false);
        }
        __syncthreads();
    }

    #pragma unroll
    for (int t = 0; t < 4; t++) {
        #pragma unroll
        for (int vr = 0; vr < 8; vr++) {
            int m  = vr + 8 * hl;
            int gr = row0 + wave * 16 + m;
            int gc = col0 + t * 16 + nn;
            float val = acc[t][vr];
            if (bias) val += bias[gc];
            if (relu) val = fmaxf(val, 0.0f);
            C[(size_t)gr * N + gc] = val;
        }
    }
}

// ---------------------------------------------------------------------------
// Fused Transformer-XL attention. One wave per (b, h, 16-query tile).
// Flash-style online softmax; rel_shift folded into a 2-tile bd band with
// in-tile shift col = n + 15 - m.
// ---------------------------------------------------------------------------
__global__ __launch_bounds__(32) void attn_wmma_kernel(
    const float* __restrict__ q, const float* __restrict__ k,
    const float* __restrict__ v, const float* __restrict__ rr,
    const float* __restrict__ wbias, const float* __restrict__ rbias,
    float* __restrict__ av)
{
    const int qt = blockIdx.x, h = blockIdx.y, b = blockIdx.z;
    const int lane = threadIdx.x;
    const int hl = lane >> 4;
    const int nn = lane & 15;
    const int i0 = qt * 16;
    const size_t base = (size_t)b * SS * DD + (size_t)h * DHH;

    __shared__ _Float16 sP[16 * 32] __attribute__((aligned(16)));   // [m][kk]
    __shared__ _Float16 sVT[64 * 32] __attribute__((aligned(16)));  // [c][kk]
    __shared__ float    sBD[16][32];                                // bd band

    // A fragments: (q + w_bias), (q + r_bias) built with packed f16 adds.
    v16h aw[2], ar[2];
    {
        const float* qrow = q + base + (size_t)(i0 + nn) * DD;
        const float* wbr  = wbias + h * DHH;
        const float* rbr  = rbias + h * DHH;
        #pragma unroll
        for (int c = 0; c < 2; c++) {
            v16h qf  = frag_ld_g(qrow, c, hl);
            v16h wbf = frag_ld_g(wbr,  c, hl);
            v16h rbf = frag_ld_g(rbr,  c, hl);
            aw[c] = qf + wbf;
            ar[c] = qf + rbf;
        }
    }

    float mrow[8], lrow[8];
    v8f O[4] = {};
    #pragma unroll
    for (int vr = 0; vr < 8; vr++) { mrow[vr] = -1e30f; lrow[vr] = 0.0f; }

    const float scale = 0.125f;  // 1/sqrt(64)
    const int nkeys = i0 + 16;

    for (int j0 = 0; j0 < nkeys; j0 += 32) {
        // ---- stage V block transposed [c][kk] into LDS (f16) --------------
        {
            const float4* vrow = (const float4*)(v + base + (size_t)(j0 + lane) * DD);
            #pragma unroll
            for (int c4 = 0; c4 < 16; c4++) {
                float4 t = vrow[c4];
                sVT[(c4 * 4 + 0) * 32 + lane] = (_Float16)t.x;
                sVT[(c4 * 4 + 1) * 32 + lane] = (_Float16)t.y;
                sVT[(c4 * 4 + 2) * 32 + lane] = (_Float16)t.z;
                sVT[(c4 * 4 + 3) * 32 + lane] = (_Float16)t.w;
            }
        }

        float sreg[2][8];

        #pragma unroll
        for (int st = 0; st < 2; st++) {
            const int jt = j0 + st * 16;
            if (jt >= nkeys) {
                #pragma unroll
                for (int vr = 0; vr < 8; vr++) sreg[st][vr] = -1e30f;
                continue;
            }
            // ac = (q+wb) @ K^T  (2 WMMAs over DH=64)
            const float* krow = k + base + (size_t)(jt + nn) * DD;
            v8f ac = {};
            #pragma unroll
            for (int c = 0; c < 2; c++) {
                v16h bK = frag_ld_g(krow, c, hl);
                ac = __builtin_amdgcn_wmma_f32_16x16x32_f16(
                    false, aw[c], false, bK, (short)0, ac, false, false);
            }
            // bd band: rr rows [t0, t0+32)
            const int t0 = jt + SS - 16 - i0;
            v8f bd1 = {}, bd2 = {};
            {
                const float* rrow = rr + base + (size_t)(t0 + nn) * DD;
                #pragma unroll
                for (int c = 0; c < 2; c++) {
                    v16h bR = frag_ld_g(rrow, c, hl);
                    bd1 = __builtin_amdgcn_wmma_f32_16x16x32_f16(
                        false, ar[c], false, bR, (short)0, bd1, false, false);
                }
            }
            if (t0 + 16 < SS) {
                const float* rrow = rr + base + (size_t)(t0 + 16 + nn) * DD;
                #pragma unroll
                for (int c = 0; c < 2; c++) {
                    v16h bR = frag_ld_g(rrow, c, hl);
                    bd2 = __builtin_amdgcn_wmma_f32_16x16x32_f16(
                        false, ar[c], false, bR, (short)0, bd2, false, false);
                }
            }
            // stash bd band, then rel-shift + causal mask + scale
            #pragma unroll
            for (int vr = 0; vr < 8; vr++) {
                int m = vr + 8 * hl;
                sBD[m][nn]      = bd1[vr];
                sBD[m][16 + nn] = bd2[vr];
            }
            asm volatile("s_wait_dscnt 0" ::: "memory");
            #pragma unroll
            for (int vr = 0; vr < 8; vr++) {
                int m = vr + 8 * hl;
                int i = i0 + m;
                int j = jt + nn;
                float bdv = sBD[m][nn + 15 - m];
                sreg[st][vr] = (j <= i) ? (ac[vr] + bdv) * scale : -1e30f;
            }
        }

        // ---- online softmax over the 32-key block -------------------------
        #pragma unroll
        for (int vr = 0; vr < 8; vr++) {
            float tmax = fmaxf(sreg[0][vr], sreg[1][vr]);
            #pragma unroll
            for (int off = 1; off < 16; off <<= 1)
                tmax = fmaxf(tmax, __shfl_xor(tmax, off, 32));
            float newm = fmaxf(mrow[vr], tmax);
            float p0 = __expf(sreg[0][vr] - newm);
            float p1 = __expf(sreg[1][vr] - newm);
            float rs = p0 + p1;
            #pragma unroll
            for (int off = 1; off < 16; off <<= 1)
                rs += __shfl_xor(rs, off, 32);
            float fac = __expf(mrow[vr] - newm);
            lrow[vr] = lrow[vr] * fac + rs;
            mrow[vr] = newm;
            #pragma unroll
            for (int t = 0; t < 4; t++) O[t][vr] *= fac;
            int m = vr + 8 * hl;
            sP[m * 32 + nn]      = (_Float16)p0;
            sP[m * 32 + 16 + nn] = (_Float16)p1;
        }
        asm volatile("s_wait_dscnt 0" ::: "memory");

        // ---- O += P @ V  (4 WMMAs: 16x16x32 over this key block) ----------
        v16h pa = frag_ld_lds(&sP[nn * 32], hl);
        #pragma unroll
        for (int t = 0; t < 4; t++) {
            v16h bv = frag_ld_lds(&sVT[(t * 16 + nn) * 32], hl);
            O[t] = __builtin_amdgcn_wmma_f32_16x16x32_f16(
                false, pa, false, bv, (short)0, O[t], false, false);
        }
    }

    // ---- normalize and store attn_vec [B,S,NH*DH] -------------------------
    #pragma unroll
    for (int vr = 0; vr < 8; vr++) {
        int m = vr + 8 * hl;
        float inv = 1.0f / lrow[vr];
        #pragma unroll
        for (int t = 0; t < 4; t++)
            av[base + (size_t)(i0 + m) * DD + t * 16 + nn] = O[t][vr] * inv;
    }
}

// ---------------------------------------------------------------------------
// out = LayerNorm(a + r) * g + beta, one row (D=1024) per block.
// ---------------------------------------------------------------------------
__global__ __launch_bounds__(256) void ln_residual_kernel(
    const float* __restrict__ a, const float* __restrict__ r,
    const float* __restrict__ g, const float* __restrict__ be,
    float* __restrict__ out)
{
    __shared__ float red[256];
    const int row = blockIdx.x;
    const int tid = threadIdx.x;
    const size_t off = (size_t)row * DD;

    float x[4];
    float s = 0.0f;
    #pragma unroll
    for (int i = 0; i < 4; i++) {
        int c = tid + i * 256;
        x[i] = a[off + c] + r[off + c];
        s += x[i];
    }
    red[tid] = s;
    __syncthreads();
    for (int w = 128; w > 0; w >>= 1) {
        if (tid < w) red[tid] += red[tid + w];
        __syncthreads();
    }
    float mu = red[0] * (1.0f / DD);
    __syncthreads();

    float vs = 0.0f;
    #pragma unroll
    for (int i = 0; i < 4; i++) {
        float d = x[i] - mu;
        vs += d * d;
    }
    red[tid] = vs;
    __syncthreads();
    for (int w = 128; w > 0; w >>= 1) {
        if (tid < w) red[tid] += red[tid + w];
        __syncthreads();
    }
    float rstd = rsqrtf(red[0] * (1.0f / DD) + 1e-5f);

    #pragma unroll
    for (int i = 0; i < 4; i++) {
        int c = tid + i * 256;
        out[off + c] = (x[i] - mu) * rstd * g[c] + be[c];
    }
}

// ---------------------------------------------------------------------------
extern "C" void kernel_launch(void* const* d_in, const int* in_sizes, int n_in,
                              void* d_out, int out_size, void* d_ws, size_t ws_size,
                              hipStream_t stream)
{
    (void)in_sizes; (void)n_in; (void)out_size; (void)ws_size;
    const float* w     = (const float*)d_in[0];
    const float* r     = (const float*)d_in[1];
    const float* wbias = (const float*)d_in[2];
    const float* rbias = (const float*)d_in[3];
    const float* Wq    = (const float*)d_in[4];
    const float* Wk    = (const float*)d_in[5];
    const float* Wv    = (const float*)d_in[6];
    const float* Wr    = (const float*)d_in[7];
    const float* Wo    = (const float*)d_in[8];
    const float* ln1g  = (const float*)d_in[9];
    const float* ln1b  = (const float*)d_in[10];
    const float* W1    = (const float*)d_in[11];
    const float* b1    = (const float*)d_in[12];
    const float* W2    = (const float*)d_in[13];
    const float* b2    = (const float*)d_in[14];
    const float* ln2g  = (const float*)d_in[15];
    const float* ln2b  = (const float*)d_in[16];
    float* out = (float*)d_out;
    float* ws  = (float*)d_ws;

    const size_t SZ = (size_t)BB * SS * DD;  // 4M floats
    float* q   = ws;
    float* kb  = ws + 1 * SZ;
    float* vb  = ws + 2 * SZ;
    float* rrb = ws + 3 * SZ;
    float* av  = ws + 4 * SZ;
    float* ao  = ws + 5 * SZ;
    float* x   = ws + 6 * SZ;
    float* ff  = ws + 7 * SZ;
    float* h1  = ws + 8 * SZ;  // 16M floats

    const int M = BB * SS;  // 4096
    dim3 blk(128);

    gemm_wmma_kernel<<<dim3(DD / 64, M / 64), blk, 0, stream>>>(w, Wq, q,  M, DD, DD, nullptr, 0);
    gemm_wmma_kernel<<<dim3(DD / 64, M / 64), blk, 0, stream>>>(w, Wk, kb, M, DD, DD, nullptr, 0);
    gemm_wmma_kernel<<<dim3(DD / 64, M / 64), blk, 0, stream>>>(w, Wv, vb, M, DD, DD, nullptr, 0);
    gemm_wmma_kernel<<<dim3(DD / 64, M / 64), blk, 0, stream>>>(r, Wr, rrb, M, DD, DD, nullptr, 0);

    attn_wmma_kernel<<<dim3(SS / 16, NHH, BB), dim3(32), 0, stream>>>(
        q, kb, vb, rrb, wbias, rbias, av);

    gemm_wmma_kernel<<<dim3(DD / 64, M / 64), blk, 0, stream>>>(av, Wo, ao, M, DD, DD, nullptr, 0);
    ln_residual_kernel<<<M, 256, 0, stream>>>(w, ao, ln1g, ln1b, x);

    gemm_wmma_kernel<<<dim3(DII / 64, M / 64), blk, 0, stream>>>(x,  W1, h1, M, DII, DD, b1, 1);
    gemm_wmma_kernel<<<dim3(DD / 64, M / 64), blk, 0, stream>>>(h1, W2, ff, M, DD, DII, b2, 0);
    ln_residual_kernel<<<M, 256, 0, stream>>>(x, ff, ln2g, ln2b, out);
}
